// QueryEncoder_3891240370278
// MI455X (gfx1250) — compile-verified, matrix operand order checked by reading, not run on hardware
//
#include <hip/hip_runtime.h>
#include <math.h>

typedef __attribute__((ext_vector_type(16))) _Float16 v16h;
typedef __attribute__((ext_vector_type(8)))  float    v8f;
typedef _Float16 h16;

#define PI_F 3.14159265358979323846f

__device__ __forceinline__ float gelu_f(float x) {
  return 0.5f * x * (1.0f + erff(x * 0.70710678118654752f));
}

// Xc = R * (q - C0); z = max(Z,1e-6); uv = XY/z*f + pp   (einsum 'cij,cnj->cni')
__device__ __forceinline__ void projpt(const float* __restrict__ R, const float* __restrict__ C0,
                                       const float* __restrict__ fo, const float* __restrict__ pp,
                                       float qx, float qy, float qz,
                                       float& u, float& v, float& z) {
  float dx = qx - C0[0], dy = qy - C0[1], dz = qz - C0[2];
  float X = R[0]*dx + R[1]*dy + R[2]*dz;
  float Y = R[3]*dx + R[4]*dy + R[5]*dz;
  float Z = R[6]*dx + R[7]*dy + R[8]*dz;
  z = Z;
  float zs = fmaxf(Z, 1e-6f);
  u = X / zs * fo[0] + pp[0];
  v = Y / zs * fo[1] + pp[1];
}

// bilinear, zero-pad taps, 3 channels, 256x256 image, channel stride 65536
__device__ __forceinline__ void sample3(const float* __restrict__ img, float x, float y, float o[3]) {
  float xf = floorf(x), yf = floorf(y);
  int x0 = (int)xf, y0 = (int)yf;
  float wx = x - xf, wy = y - yf;
  o[0] = 0.f; o[1] = 0.f; o[2] = 0.f;
#pragma unroll
  for (int dy = 0; dy < 2; ++dy) {
#pragma unroll
    for (int dx = 0; dx < 2; ++dx) {
      int xx = x0 + dx, yy = y0 + dy;
      if (xx < 0 || xx >= 256 || yy < 0 || yy >= 256) continue;
      float w = (dx ? wx : 1.f - wx) * (dy ? wy : 1.f - wy);
      const float* p = img + yy * 256 + xx;
      o[0] += w * p[0];
      o[1] += w * p[65536];
      o[2] += w * p[131072];
    }
  }
}

// ---------------------------------------------------------------------------
// WMMA GEMM:  C[m,n] = act( sum_k A[m,k]*W[n,k] + bias[n] )
// A: f16 (M x Kp), W: f16 packed (Np x Kp, Np >= roundup(N,64) rows when
// N < 64, zero padded), Kp % 32 == 0, M % 64 == 0.
//
// Block = 4 waves = 64(M) x 64(N) C tile. Per K-step:
//   - 64x32-half B tile cooperatively staged in LDS (row stride 40 halves =
//     80 B: 16B-aligned for b128, spreads banks),
//   - each wave loads one A fragment (2 x b128) and reuses it for 4 WMMAs
//     against 4 B fragments read from LDS (ds_load_b128).
// act: 0=none, 1=gelu(exact), 2=sigmoid
// ---------------------------------------------------------------------------
#define BS_STRIDE 40   // halves per LDS row (32 payload + 8 pad)

__global__ void k_gemm(const h16* __restrict__ A, const h16* __restrict__ Wp,
                       const float* __restrict__ bias,
                       float* outF, int ldF, h16* outH, int ldH,
                       int M, int N, int Kp, int act) {
  __shared__ __align__(16) h16 Bs[64 * BS_STRIDE];

  int lane = threadIdx.x;
  int tid  = threadIdx.y * 32 + lane;   // 0..127
  int g = lane >> 4;                    // lane group (K-half select)
  int r = lane & 15;                    // row (A) / col (B) index within tile
  int mTile = blockIdx.y * 4 + threadIdx.y;   // grids launched exact: always valid
  int nBase = blockIdx.x * 64;

  const h16* aRow = A + (size_t)(mTile * 16 + r) * Kp + 8 * g;

  // cooperative B stager: 128 threads x 16 halves = 64 rows x 32 halves
  int ldRow   = tid >> 1;          // 0..63
  int ldHalf  = (tid & 1) * 16;    // 0 or 16
  const h16* bSrc = Wp + (size_t)(nBase + ldRow) * Kp + ldHalf;
  h16*       bDst = Bs + ldRow * BS_STRIDE + ldHalf;

  v8f acc[4] = {};
  union Frag { v16h h; uint4 q[2]; };

  for (int kb = 0; kb < Kp; kb += 32) {
    __syncthreads();   // protect previous iteration's Bs reads
    {
      uint4 c0 = *(const uint4*)(bSrc + kb);
      uint4 c1 = *(const uint4*)(bSrc + kb + 8);
      *(uint4*)(bDst)     = c0;
      *(uint4*)(bDst + 8) = c1;
    }
    Frag a;
    a.q[0] = *(const uint4*)(aRow + kb);
    a.q[1] = *(const uint4*)(aRow + kb + 16);
    __syncthreads();   // Bs tile ready
#pragma unroll
    for (int nn = 0; nn < 4; ++nn) {
      const h16* bs = Bs + (nn * 16 + r) * BS_STRIDE + 8 * g;
      Frag b;
      b.q[0] = *(const uint4*)(bs);
      b.q[1] = *(const uint4*)(bs + 16);
      acc[nn] = __builtin_amdgcn_wmma_f32_16x16x32_f16(false, a.h, false, b.h,
                                                       (short)0, acc[nn], false, false);
    }
  }

#pragma unroll
  for (int nn = 0; nn < 4; ++nn) {
    int n = nBase + nn * 16 + r;     // C layout: col = lane%16
    if (n < N) {
      float bv = bias[n];
#pragma unroll
      for (int rr = 0; rr < 8; ++rr) {
        int m = mTile * 16 + rr + 8 * g;   // C layout: row = vgpr + 8*(lane/16)
        float val = acc[nn][rr] + bv;
        if (act == 1)      val = gelu_f(val);
        else if (act == 2) val = 1.f / (1.f + expf(-val));
        if (outF) outF[(size_t)m * ldF + n] = val;
        if (outH) outH[(size_t)m * ldH + n] = (h16)val;
      }
    }
  }
}

// f32 (N,K) weights -> f16 (Np,Kp) zero-padded
__global__ void k_pack(const float* __restrict__ w, h16* __restrict__ o,
                       int N, int K, int Np, int Kp) {
  int gid = blockIdx.x * blockDim.x + threadIdx.x;
  if (gid >= Np * Kp) return;
  int np = gid / Kp, kp = gid % Kp;
  o[gid] = (np < N && kp < K) ? (h16)w[np * K + kp] : (h16)0.f;
}

// ---------------------------------------------------------------------------
// projection + vis + fourier(pos 40->pad64) + fourier(depth 20->pad32)
// thread = (b,k,cam) row, 4096 total
// ---------------------------------------------------------------------------
__global__ void k_project(const float* __restrict__ R, const float* __restrict__ C0,
                          const float* __restrict__ fo, const float* __restrict__ pp,
                          const float* __restrict__ q, const float* __restrict__ cube_scale,
                          const float* __restrict__ dns,
                          float* __restrict__ p2d, int* __restrict__ visI,
                          float* __restrict__ visOut,
                          h16* __restrict__ Apos, h16* __restrict__ Adep) {
  int idx = blockIdx.x * blockDim.x + threadIdx.x;
  if (idx >= 4096) return;
  int bk = idx >> 2, c = idx & 3;
  float qx = q[bk * 3 + 0], qy = q[bk * 3 + 1], qz = q[bk * 3 + 2];
  const float* Rc = R + c * 9;  const float* Cc = C0 + c * 3;
  const float* fc = fo + c * 2; const float* pc = pp + c * 2;
  float u, v, z;
  projpt(Rc, Cc, fc, pc, qx, qy, qz, u, v, z);
  p2d[(c * 512 + bk) * 2 + 0] = u;   // layout (cam, b, k, 2)
  p2d[(c * 512 + bk) * 2 + 1] = v;
  int vis = (z > 0.f) && (u >= 2.f) && (u <= 254.f) && (v >= 2.f) && (v <= 254.f);
  visI[idx] = vis;
  visOut[idx] = vis ? 1.f : 0.f;

  float un = u * (1.f / 256.f), vn = v * (1.f / 256.f);
  h16* ap = Apos + (size_t)idx * 64;
#pragma unroll
  for (int j = 0; j < 10; ++j) {
    float fr = exp2f((float)j) * PI_F;
    ap[j]      = (h16)sinf(un * fr);
    ap[10 + j] = (h16)cosf(un * fr);
    ap[20 + j] = (h16)sinf(vn * fr);
    ap[30 + j] = (h16)cosf(vn * fr);
  }
#pragma unroll
  for (int j = 40; j < 64; ++j) ap[j] = (h16)0.f;

  float dx = qx - Cc[0], dy = qy - Cc[1], dz = qz - Cc[2];
  float dist = sqrtf(dx * dx + dy * dy + dz * dz) / (cube_scale[0] * dns[0]);
  h16* dp = Adep + (size_t)idx * 32;
#pragma unroll
  for (int j = 0; j < 10; ++j) {
    float fr = exp2f((float)j) * PI_F;
    dp[j]      = (h16)sinf(dist * fr);
    dp[10 + j] = (h16)cosf(dist * fr);
  }
#pragma unroll
  for (int j = 20; j < 32; ++j) dp[j] = (h16)0.f;
}

// ---------------------------------------------------------------------------
// cube sampling: masked mean over cams, written in conv3d layout (C, d=kz, h=j, w=i)
// ---------------------------------------------------------------------------
__global__ void k_cube(const float* __restrict__ views, const float* __restrict__ R,
                       const float* __restrict__ C0, const float* __restrict__ fo,
                       const float* __restrict__ pp, const float* __restrict__ q,
                       const int* __restrict__ qt, const float* __restrict__ cube_scale,
                       float* __restrict__ mv) {
  int gid = blockIdx.x * blockDim.x + threadIdx.x;
  if (gid >= 512 * 125) return;
  int bk = gid / 125, p = gid % 125;
  int i = p / 25, j = (p % 25) / 5, kz = p % 5;   // xyz flatten order i*25+j*5+k
  int b = bk >> 8;
  float s2 = cube_scale[0] * 2.f;
  float qx = q[bk * 3 + 0] + (i - 2) * s2;
  float qy = q[bk * 3 + 1] + (j - 2) * s2;
  float qz = q[bk * 3 + 2] + (kz - 2) * s2;
  int t = qt[bk];
  float sum0 = 0.f, sum1 = 0.f, sum2 = 0.f, cnt = 0.f;
  for (int c = 0; c < 4; ++c) {
    float u, v, z;
    projpt(R + c * 9, C0 + c * 3, fo + c * 2, pp + c * 2, qx, qy, qz, u, v, z);
    bool m = (u >= 0.f) && (u <= 256.f) && (v >= 0.f) && (v <= 256.f);
    if (m) {
      const float* img = views + (size_t)(((c * 2 + b) * 16 + t) * 3) * 65536;
      float s[3];
      sample3(img, u - 0.5f, v - 0.5f, s);
      sum0 += s[0]; sum1 += s[1]; sum2 += s[2]; cnt += 1.f;
    }
  }
  cnt = fmaxf(cnt, 1.f);
  float* o = mv + (size_t)bk * 375;
  int vox = kz * 25 + j * 5 + i;   // (d,h,w) = (kz, j, i)
  o[0 * 125 + vox] = sum0 / cnt;
  o[1 * 125 + vox] = sum1 / cnt;
  o[2 * 125 + vox] = sum2 / cnt;
}

// ---------------------------------------------------------------------------
// 3x3x3 conv (3->8) SAME + gelu, output transposed to lin_vol flatten order
// col = v*125 + w*25 + h*5 + d ;  Kp padded to 1024 (cols >=1000 zeroed)
// ---------------------------------------------------------------------------
__global__ void k_conv3d(const float* __restrict__ mv, const float* __restrict__ w,
                         const float* __restrict__ bias, h16* __restrict__ Avol) {
  int gid = blockIdx.x * blockDim.x + threadIdx.x;
  if (gid >= 512 * 1024) return;
  int bk = gid >> 10, cp = gid & 1023;
  if (cp >= 1000) { Avol[gid] = (h16)0.f; return; }
  int v = cp / 125, rem = cp % 125;
  int ix = rem / 25, jy = (rem % 25) / 5, dz = rem % 5;
  const float* m = mv + (size_t)bk * 375;
  float acc = bias[v];
  for (int ic = 0; ic < 3; ++ic)
    for (int kd = 0; kd < 3; ++kd) {
      int d = dz + kd - 1; if (d < 0 || d > 4) continue;
      for (int kh = 0; kh < 3; ++kh) {
        int h = jy + kh - 1; if (h < 0 || h > 4) continue;
        for (int kw = 0; kw < 3; ++kw) {
          int ww = ix + kw - 1; if (ww < 0 || ww > 4) continue;
          acc += m[ic * 125 + d * 25 + h * 5 + ww] *
                 w[(((v * 3 + ic) * 3 + kd) * 3 + kh) * 3 + kw];
        }
      }
    }
  Avol[gid] = (h16)gelu_f(acc);
}

// ---------------------------------------------------------------------------
// 9x9 patch sampling around p2d (pixel coords = uv + offset exactly)
// rows ordered (cam, b, k); patch layout (ch, ry, cx)
// ---------------------------------------------------------------------------
__global__ void k_patch(const float* __restrict__ views, const float* __restrict__ p2d,
                        const int* __restrict__ qt, float* __restrict__ patches) {
  int gid = blockIdx.x * blockDim.x + threadIdx.x;
  if (gid >= 2048 * 81) return;
  int row = gid / 81, pix = gid % 81;
  int ry = pix / 9, cx = pix % 9;
  int c = row / 512, bk = row % 512, b = bk >> 8;
  int t = qt[bk];
  float u = p2d[row * 2 + 0], v = p2d[row * 2 + 1];
  const float* img = views + (size_t)(((c * 2 + b) * 16 + t) * 3) * 65536;
  float s[3];
  sample3(img, u + (float)(cx - 4), v + (float)(ry - 4), s);
  float* o = patches + (size_t)row * 243;
  o[0 * 81 + pix] = s[0];
  o[1 * 81 + pix] = s[1];
  o[2 * 81 + pix] = s[2];
}

// 3x3 SAME conv on 9x9 + gelu; f16 output (feeds WMMA)
template <typename Tin>
__global__ void k_conv2d(const Tin* __restrict__ in, const float* __restrict__ w,
                         const float* __restrict__ bias, h16* __restrict__ out,
                         int Ci, int Co) {
  int gid = blockIdx.x * blockDim.x + threadIdx.x;
  int total = 2048 * Co * 81;
  if (gid >= total) return;
  int row = gid / (Co * 81);
  int r2 = gid % (Co * 81);
  int oc = r2 / 81, pix = r2 % 81;
  int ry = pix / 9, cx = pix % 9;
  const Tin* ib = in + (size_t)row * Ci * 81;
  float acc = bias[oc];
  for (int ic = 0; ic < Ci; ++ic) {
    const float* wk = w + ((size_t)oc * Ci + ic) * 9;
    const Tin* ip = ib + ic * 81;
    for (int ky = 0; ky < 3; ++ky) {
      int iy = ry + ky - 1; if (iy < 0 || iy > 8) continue;
      for (int kx = 0; kx < 3; ++kx) {
        int ixx = cx + kx - 1; if (ixx < 0 || ixx > 8) continue;
        acc += (float)ip[iy * 9 + ixx] * wk[ky * 3 + kx];
      }
    }
  }
  out[(size_t)row * (Co * 81) + oc * 81 + pix] = (h16)gelu_f(acc);
}

// fill terms[row, 256..1535]: volume (broadcast over cams), patch (reordered),
// qt/tt/vis embedding lookups. pos (cols 0..255) / depth (1536..1791) are
// written directly by their GEMMs.
__global__ void k_assemble(h16* __restrict__ terms, const h16* __restrict__ volE,
                           const h16* __restrict__ embP, const float* __restrict__ tq,
                           const float* __restrict__ tte, const float* __restrict__ vtab,
                           const int* __restrict__ visI, const int* __restrict__ qt,
                           const int* __restrict__ tgt) {
  int gid = blockIdx.x * blockDim.x + threadIdx.x;
  if (gid >= 4096 * 256) return;
  int row = gid >> 8, e = gid & 255;
  int c = row & 3, bk = row >> 2;
  h16* tb = terms + (size_t)row * 1792;
  tb[256 + e]  = volE[(size_t)bk * 256 + e];
  tb[512 + e]  = embP[((size_t)c * 512 + bk) * 256 + e];
  tb[768 + e]  = (h16)tq[qt[bk] * 256 + e];
  tb[1024 + e] = (h16)tte[tgt[bk] * 256 + e];
  tb[1280 + e] = (h16)vtab[visI[row] * 256 + e];
}

// fused[row,e] = sum_i gates[row,i] * terms[row, i*256+e]
__global__ void k_fuse(const float* __restrict__ gates, const h16* __restrict__ terms,
                       h16* __restrict__ fusedA) {
  int gid = blockIdx.x * blockDim.x + threadIdx.x;
  if (gid >= 4096 * 256) return;
  int row = gid >> 8, e = gid & 255;
  const h16* tb = terms + (size_t)row * 1792;
  const float* gb = gates + (size_t)row * 8;
  float f = 0.f;
#pragma unroll
  for (int i = 0; i < 7; ++i) f += gb[i] * (float)tb[i * 256 + e];
  fusedA[gid] = (h16)f;
}

// ---------------------------------------------------------------------------
extern "C" void kernel_launch(void* const* d_in, const int* in_sizes, int n_in,
                              void* d_out, int out_size, void* d_ws, size_t ws_size,
                              hipStream_t stream) {
  const float* views       = (const float*)d_in[0];
  const float* camR        = (const float*)d_in[1];
  const float* camC        = (const float*)d_in[2];
  const float* camF        = (const float*)d_in[3];
  const float* camP        = (const float*)d_in[4];
  const float* qc          = (const float*)d_in[5];
  const int*   qtime       = (const int*)d_in[6];
  const int*   ttime       = (const int*)d_in[7];
  const float* cscale      = (const float*)d_in[8];
  const float* v2v_w       = (const float*)d_in[9];
  const float* v2v_b       = (const float*)d_in[10];
  const float* lin_vol_w   = (const float*)d_in[11];
  const float* lin_vol_b   = (const float*)d_in[12];
  const float* lin_pos_w   = (const float*)d_in[13];
  const float* lin_pos_b   = (const float*)d_in[14];
  const float* lin_depth_w = (const float*)d_in[15];
  const float* lin_depth_b = (const float*)d_in[16];
  const float* c1w = (const float*)d_in[17];
  const float* c1b = (const float*)d_in[18];
  const float* c2w = (const float*)d_in[19];
  const float* c2b = (const float*)d_in[20];
  const float* c3w = (const float*)d_in[21];
  const float* c3b = (const float*)d_in[22];
  const float* m1w = (const float*)d_in[23];
  const float* m1b = (const float*)d_in[24];
  const float* m2w = (const float*)d_in[25];
  const float* m2b = (const float*)d_in[26];
  const float* tq_emb = (const float*)d_in[27];
  const float* tt_emb = (const float*)d_in[28];
  const float* vtab   = (const float*)d_in[29];
  const float* dns    = (const float*)d_in[30];
  const float* gate_w = (const float*)d_in[31];
  const float* gate_b = (const float*)d_in[32];
  const float* f1w = (const float*)d_in[33];
  const float* f1b = (const float*)d_in[34];
  const float* f2w = (const float*)d_in[35];
  const float* f2b = (const float*)d_in[36];
  (void)in_sizes; (void)n_in; (void)out_size; (void)ws_size;

  float* out    = (float*)d_out;            // embed_total: (B,K,ncams,E) = 4096 x 256
  float* visOut = out + (size_t)4096 * 256; // vis as float: 4096

  // ---- workspace bump allocator (256B aligned for b128 fragment loads) ----
  char* wsBase = (char*)d_ws;
  size_t off = 0;
  auto alloc = [&](size_t bytes) -> void* {
    void* p = wsBase + off;
    off += (bytes + 255) & ~(size_t)255;
    return p;
  };
  float* p2d     = (float*)alloc((size_t)2048 * 2 * 4);
  int*   visI    = (int*)  alloc((size_t)4096 * 4);
  h16*   Apos    = (h16*)  alloc((size_t)4096 * 64 * 2);
  h16*   Adep    = (h16*)  alloc((size_t)4096 * 32 * 2);
  h16*   terms   = (h16*)  alloc((size_t)4096 * 1792 * 2);
  float* mv      = (float*)alloc((size_t)512 * 375 * 4);
  h16*   Avol    = (h16*)  alloc((size_t)512 * 1024 * 2);
  h16*   volE    = (h16*)  alloc((size_t)512 * 256 * 2);
  float* patches = (float*)alloc((size_t)2048 * 243 * 4);
  h16*   c1o     = (h16*)  alloc((size_t)2048 * 32 * 81 * 2);
  h16*   c2o     = (h16*)  alloc((size_t)2048 * 64 * 81 * 2);
  h16*   Apatch  = (h16*)  alloc((size_t)2048 * 10368 * 2);
  h16*   m1o     = (h16*)  alloc((size_t)2048 * 512 * 2);
  h16*   embP    = (h16*)  alloc((size_t)2048 * 256 * 2);
  float* gates   = (float*)alloc((size_t)4096 * 8 * 4);
  h16*   fusedA  = (h16*)  alloc((size_t)4096 * 256 * 2);
  h16*   hbuf    = (h16*)  alloc((size_t)4096 * 1024 * 2);
  h16*   Wpos    = (h16*)  alloc((size_t)256 * 64 * 2);
  h16*   Wdep    = (h16*)  alloc((size_t)256 * 32 * 2);
  h16*   Wvol    = (h16*)  alloc((size_t)256 * 1024 * 2);
  h16*   Wm1     = (h16*)  alloc((size_t)512 * 10368 * 2);
  h16*   Wm2     = (h16*)  alloc((size_t)256 * 512 * 2);
  h16*   Wgate   = (h16*)  alloc((size_t)64 * 1792 * 2);   // padded to 64 rows (N-tile)
  h16*   Wf1     = (h16*)  alloc((size_t)1024 * 256 * 2);
  h16*   Wf2     = (h16*)  alloc((size_t)256 * 1024 * 2);

  auto cdiv = [](int a, int b) { return (a + b - 1) / b; };
  auto gemm = [&](const h16* A, const h16* Wp, const float* bias,
                  float* oF, int ldF, h16* oH, int ldH,
                  int M, int N, int Kp, int act) {
    dim3 grid(cdiv(N, 64), M / 64);   // all M here are multiples of 64
    dim3 blk(32, 4);
    k_gemm<<<grid, blk, 0, stream>>>(A, Wp, bias, oF, ldF, oH, ldH, M, N, Kp, act);
  };

  // ---- pack all linear weights to f16 with K padded to multiples of 32 ----
  k_pack<<<cdiv(256 * 64, 256), 256, 0, stream>>>(lin_pos_w,   Wpos, 256, 40,   256, 64);
  k_pack<<<cdiv(256 * 32, 256), 256, 0, stream>>>(lin_depth_w, Wdep, 256, 20,   256, 32);
  k_pack<<<cdiv(256 * 1024, 256), 256, 0, stream>>>(lin_vol_w, Wvol, 256, 1000, 256, 1024);
  k_pack<<<cdiv(512 * 10368, 256), 256, 0, stream>>>(m1w,      Wm1,  512, 10368, 512, 10368);
  k_pack<<<cdiv(256 * 512, 256), 256, 0, stream>>>(m2w,        Wm2,  256, 512,  256, 512);
  k_pack<<<cdiv(64 * 1792, 256), 256, 0, stream>>>(gate_w,     Wgate, 7,  1792, 64,  1792);
  k_pack<<<cdiv(1024 * 256, 256), 256, 0, stream>>>(f1w,       Wf1, 1024, 256, 1024, 256);
  k_pack<<<cdiv(256 * 1024, 256), 256, 0, stream>>>(f2w,       Wf2,  256, 1024, 256, 1024);

  // ---- projection / fourier features / vis ----
  k_project<<<cdiv(4096, 256), 256, 0, stream>>>(camR, camC, camF, camP, qc, cscale, dns,
                                                 p2d, visI, visOut, Apos, Adep);
  // embed_pos -> terms[:,0:256], embed_depth -> terms[:,1536:1792]
  gemm(Apos, Wpos, lin_pos_b,   nullptr, 0, terms + 0,    1792, 4096, 256, 64, 0);
  gemm(Adep, Wdep, lin_depth_b, nullptr, 0, terms + 1536, 1792, 4096, 256, 32, 0);

  // ---- volume branch ----
  k_cube<<<cdiv(512 * 125, 256), 256, 0, stream>>>(views, camR, camC, camF, camP, qc,
                                                   qtime, cscale, mv);
  k_conv3d<<<cdiv(512 * 1024, 256), 256, 0, stream>>>(mv, v2v_w, v2v_b, Avol);
  gemm(Avol, Wvol, lin_vol_b, nullptr, 0, volE, 256, 512, 256, 1024, 0);

  // ---- patch branch ----
  k_patch<<<cdiv(2048 * 81, 256), 256, 0, stream>>>(views, p2d, qtime, patches);
  k_conv2d<float><<<cdiv(2048 * 32 * 81, 256), 256, 0, stream>>>(patches, c1w, c1b, c1o, 3, 32);
  k_conv2d<h16><<<cdiv(2048 * 64 * 81, 256), 256, 0, stream>>>(c1o, c2w, c2b, c2o, 32, 64);
  k_conv2d<h16><<<cdiv(2048 * 128 * 81, 256), 256, 0, stream>>>(c2o, c3w, c3b, Apatch, 64, 128);
  gemm(Apatch, Wm1, m1b, nullptr, 0, m1o, 512, 2048, 512, 10368, 1);  // + gelu
  gemm(m1o,    Wm2, m2b, nullptr, 0, embP, 256, 2048, 256, 512, 0);

  // ---- gated fusion ----
  k_assemble<<<cdiv(4096 * 256, 256), 256, 0, stream>>>(terms, volE, embP, tq_emb, tt_emb,
                                                        vtab, visI, qtime, ttime);
  gemm(terms, Wgate, gate_b, gates, 8, nullptr, 0, 4096, 7, 1792, 2);  // sigmoid
  k_fuse<<<cdiv(4096 * 256, 256), 256, 0, stream>>>(gates, terms, fusedA);
  gemm(fusedA, Wf1, f1b, nullptr, 0, hbuf, 1024, 4096, 1024, 256, 1);  // + gelu
  gemm(hbuf,   Wf2, f2b, out, 256, nullptr, 0, 4096, 256, 1024, 0);    // -> d_out
}
